// ReactionCenterPredictor_29557964931690
// MI455X (gfx1250) — matrix operand order repaired and checked by reading, not compiled, not run on hardware
//
#include <hip/hip_runtime.h>

// CDNA5 / gfx1250 WMMA types: wave32, 16x16x32 f16 -> f32 accum
typedef __attribute__((ext_vector_type(16))) _Float16 v16h;
typedef __attribute__((ext_vector_type(8)))  float    v8f;

union V16H {
    v16h v;
    _Float16 e[16];
};

__device__ __forceinline__ float sigmoid_f(float x) {
    return 1.0f / (1.0f + __expf(-x));
}

// One 16-edge tile, fully in the matrix pipe:
//   WMMA#1: H = W1(16x32 f16, A) x X^T(32x16 f16, B)  -> lane (p,hi) vgpr r = h_pre[hidden hi*8+r][edge p]
//   epilogue: + b1 + energy*W1[:,32], relu, pack to f16 (K = hi*16 + 0..7)
//   WMMA#2: logits = w2row(A, nonzero in elements 0..7 of both lane halves) x relu(H)(B) -> every vgpr = logit[edge p]
// Returns logit of edge (edgeBase + p); identical on both lane halves.
// All indexing is 32-bit: table extents (128MB emb, 64MB idx) fit in u32 byte offsets,
// so gathers lower to SADDR + 32-bit VGPR offset (GVS) addressing.
__device__ __forceinline__ float tile_logit(
    const float* __restrict__ emb,
    const float* __restrict__ energies,
    const int* __restrict__ eidx32,      // low dwords of int64 edge_index (values < N)
    int E,
    v16h aW1, v16h aW2,
    const float (&b1r)[8], const float (&w1l)[8],
    float we, float bee,
    int edgeBase, int edgeEnd,
    int p, int hi)
{
    // --- B1 gather: lane (p,hi) supplies column p = edge p's features K = hi*16 + 0..15
    int m = edgeBase + p;
    if (m >= edgeEnd) m = edgeEnd - 1;                  // clamp (tail safety)
    int s = eidx32[2 * m];                              // src atom (low dword of int64)
    int d = eidx32[2 * (E + m)];                        // dst atom
    const float4* ps = (const float4*)(emb + s * 32 + hi * 16);
    const float4* pd = (const float4*)(emb + d * 32 + hi * 16);
    float4 s0 = ps[0], s1 = ps[1], s2 = ps[2], s3 = ps[3];   // contiguous 64B slice
    float4 d0 = pd[0], d1 = pd[1], d2 = pd[2], d3 = pd[3];

    V16H x;
    x.e[0]  = (_Float16)(s0.x + d0.x);  x.e[1]  = (_Float16)(s0.y + d0.y);
    x.e[2]  = (_Float16)(s0.z + d0.z);  x.e[3]  = (_Float16)(s0.w + d0.w);
    x.e[4]  = (_Float16)(s1.x + d1.x);  x.e[5]  = (_Float16)(s1.y + d1.y);
    x.e[6]  = (_Float16)(s1.z + d1.z);  x.e[7]  = (_Float16)(s1.w + d1.w);
    x.e[8]  = (_Float16)(s2.x + d2.x);  x.e[9]  = (_Float16)(s2.y + d2.y);
    x.e[10] = (_Float16)(s2.z + d2.z);  x.e[11] = (_Float16)(s2.w + d2.w);
    x.e[12] = (_Float16)(s3.x + d3.x);  x.e[13] = (_Float16)(s3.y + d3.y);
    x.e[14] = (_Float16)(s3.z + d3.z);  x.e[15] = (_Float16)(s3.w + d3.w);

    float en = energies[m] * we + bee;                  // this lane's edge energy feature

    // --- WMMA#1: h_pre = W1 x X^T
    v8f c = {};
    c = __builtin_amdgcn_wmma_f32_16x16x32_f16(
            false, aW1, false, x.v, (short)0, c, false, false);

    // --- epilogue: bias + energy column + relu, pack as B matrix for WMMA#2
    V16H h2;
#pragma unroll
    for (int r = 0; r < 8; ++r) {
        float h = c[r] + b1r[r] + en * w1l[r];
        h = fmaxf(h, 0.0f);
        h2.e[r]     = (_Float16)h;                      // K = hi*16 + r (hidden hi*8+r)
        h2.e[8 + r] = (_Float16)0.0f;                   // zero pad K = hi*16 + 8..15
    }

    // --- WMMA#2: logits = w2 x relu(H); every accumulator vgpr = logit[edge p]
    v8f c2 = {};
    c2 = __builtin_amdgcn_wmma_f32_16x16x32_f16(
            false, aW2, false, h2.v, (short)0, c2, false, false);
    return c2[0];
}

__global__ __launch_bounds__(256)
void ReactionCenterPredictor_29557964931690_kernel(
    const float* __restrict__ emb,       // [N,32]
    const float* __restrict__ energies,  // [E]
    const float* __restrict__ We,        // [1,1]
    const float* __restrict__ be,        // [1]
    const float* __restrict__ W1,        // [16,33]
    const float* __restrict__ b1,        // [16]
    const float* __restrict__ W2,        // [1,16]
    const float* __restrict__ b2,        // [1]
    const int* __restrict__ eidx32,      // [2,E] int64 viewed as int32 pairs
    float* __restrict__ out,             // [E/2]
    int E, int halfE, int numTiles)
{
    const int lane = threadIdx.x & 31;
    const int p    = lane & 15;
    const int hi   = lane >> 4;

    int tile = blockIdx.x * (blockDim.x >> 5) + (threadIdx.x >> 5);
    if (tile >= numTiles) return;                       // wave-uniform: EXEC stays all-1s
    int base = tile * 16;

    // A1 = W1 (16 hidden x 32 features), A-layout: lane p = hidden row p,
    // element e -> K=hi*16+e
    V16H aW1;
    const float* w1p = W1 + p * 33;
#pragma unroll
    for (int i = 0; i < 8; ++i) {
        aW1.e[i]     = (_Float16)w1p[hi * 16 + i];
        aW1.e[8 + i] = (_Float16)w1p[hi * 16 + 8 + i];
    }

    // A2 = w2 broadcast matching relu(H)'s K placement:
    // nonzero K are {0..7, 16..23} -> element e<8 = w2[hi*8+e], e>=8 zero
    V16H aW2;
#pragma unroll
    for (int i = 0; i < 16; ++i)
        aW2.e[i] = (i < 8) ? (_Float16)W2[hi * 8 + i] : (_Float16)0.0f;

    // per-lane hidden-slice constants: bias and the K=32 (energy) weight column
    float b1r[8], w1l[8];
#pragma unroll
    for (int r = 0; r < 8; ++r) {
        int hh = hi * 8 + r;
        b1r[r] = b1[hh];
        w1l[r] = W1[hh * 33 + 32];
    }
    const float we  = We[0];
    const float bee = be[0];
    const float b2v = b2[0];

    // prefetch next tile's gather indices (global_prefetch_b8)
    if (base + 16 < halfE) {
        __builtin_prefetch(eidx32 + 2 * (base + 16), 0, 1);
        __builtin_prefetch(eidx32 + 2 * (E + base + 16), 0, 1);
    }

    // lower-half tile and its paired upper-half tile
    float lgL = tile_logit(emb, energies, eidx32, E, aW1.v, aW2.v, b1r, w1l, we, bee,
                           base,         halfE, p, hi);
    float lgH = tile_logit(emb, energies, eidx32, E, aW1.v, aW2.v, b1r, w1l, we, bee,
                           halfE + base, E,     p, hi);

    float sc = 0.5f * (sigmoid_f(lgL + b2v) + sigmoid_f(lgH + b2v));

    // lanes 0..15 store 16 consecutive outputs (contiguous 64B)
    if (hi == 0) {
        int oi = base + p;
        if (oi < halfE) out[oi] = sc;
    }
}

extern "C" void kernel_launch(void* const* d_in, const int* in_sizes, int n_in,
                              void* d_out, int out_size, void* d_ws, size_t ws_size,
                              hipStream_t stream) {
    (void)n_in; (void)d_ws; (void)ws_size;
    const float* emb      = (const float*)d_in[0];
    const float* energies = (const float*)d_in[1];
    const float* We       = (const float*)d_in[2];
    const float* be       = (const float*)d_in[3];
    const float* W1       = (const float*)d_in[4];
    const float* b1       = (const float*)d_in[5];
    const float* W2       = (const float*)d_in[6];
    const float* b2       = (const float*)d_in[7];
    const int*   eidx32   = (const int*)d_in[8];        // int64 indices, low dwords used
    float*       out      = (float*)d_out;

    int E        = in_sizes[1];                         // bond_energies has E elements
    int halfE    = out_size;                            // E/2 outputs
    int numTiles = (halfE + 15) / 16;                   // 16 output pairs per wave
    const int threads = 256;                            // 8 wave32 per block
    int blocks = (numTiles + 7) / 8;
    if (blocks < 1) blocks = 1;

    ReactionCenterPredictor_29557964931690_kernel<<<blocks, threads, 0, stream>>>(
        emb, energies, We, be, W1, b1, W2, b2, eidx32, out, E, halfE, numTiles);
}
